// Attention_369367187796
// MI455X (gfx1250) — matrix operand order repaired and checked by reading, not compiled, not run on hardware
//
#include <hip/hip_runtime.h>
#include <stdint.h>

// ---------------------------------------------------------------------------
// Attention: Q=Wq@x, K=Wk@x, V=Wv@x, S=Q@K^T (over n), P=softmax(S/32, axis=q),
// out = P@V.   B=4, DX=DQ=DK=DV=1024, N=4096.
// All matmuls: v_wmma_f32_16x16x32_bf16; LDS staged with
// global_load_async_to_lds_b128 (ASYNCcnt) + double buffering.
// ---------------------------------------------------------------------------

typedef __bf16 v16bf __attribute__((ext_vector_type(16)));
typedef float  v8f   __attribute__((ext_vector_type(8)));

struct __align__(16) U4 { unsigned int x, y, z, w; };
struct __align__(8)  F2 { float x, y; };

union ABFrag {
  U4 q[2];
  unsigned int d[8];
  v16bf v;
};

__device__ __forceinline__ unsigned short f2bf(float f) {
  unsigned int u = __float_as_uint(f);
  u += 0x7fffu + ((u >> 16) & 1u);            // round-to-nearest-even
  return (unsigned short)(u >> 16);
}
__device__ __forceinline__ unsigned int pack2(float a, float b) {
  return (unsigned int)f2bf(a) | ((unsigned int)f2bf(b) << 16);
}

// LDS byte offset of a __shared__ object: low 32 bits of its flat address.
__device__ __forceinline__ unsigned lds_off(const void* p) {
  return (unsigned)(size_t)p;
}
// Async DMA global->LDS, 64 contiguous bytes per lane (4 x b128).
// INST_OFFSET applies to both the LDS and global address, so one address pair
// serves all four chunks.
__device__ __forceinline__ void async_copy64B(unsigned lds, const void* g) {
  asm volatile(
      "global_load_async_to_lds_b128 %0, %1, off\n\t"
      "global_load_async_to_lds_b128 %0, %1, off offset:16\n\t"
      "global_load_async_to_lds_b128 %0, %1, off offset:32\n\t"
      "global_load_async_to_lds_b128 %0, %1, off offset:48"
      :: "v"(lds), "v"((unsigned long long)(size_t)g) : "memory");
}
__device__ __forceinline__ void wait_async0() {
  asm volatile("s_wait_asynccnt 0x0" ::: "memory");
}

#define BM 128
#define BN 128
#define BK 64
#define LDA 72   // bf16 elems per LDS row (144B: keeps 16B alignment)

// ---------------------------------------------------------------------------
// f32 -> bf16 plain (row-major), 8 floats -> 4 dwords per thread
// ---------------------------------------------------------------------------
__global__ __launch_bounds__(256) void cvt_plain(
    const float* __restrict__ src, unsigned int* __restrict__ dst)
{
  const size_t i = ((size_t)blockIdx.x * 256 + threadIdx.x) * 4;  // dword index
  const F2* s = (const F2*)src + i;
  #pragma unroll
  for (int j = 0; j < 4; ++j) dst[i + j] = pack2(s[j].x, s[j].y);
}

// ---------------------------------------------------------------------------
// f32 (rows, Nn) -> bf16 K-pair-packed: dst dword(p, n) = (src[2p][n], src[2p+1][n])
// ---------------------------------------------------------------------------
__global__ __launch_bounds__(256) void cvt_pack_pairs(
    const float* __restrict__ src, unsigned int* __restrict__ dst, int Nn)
{
  const int gid = blockIdx.x * 256 + threadIdx.x;
  const int pr  = gid / (Nn / 8);
  const int n   = (gid % (Nn / 8)) * 8;
  const float* r0 = src + (size_t)(2 * pr) * Nn + n;
  const float* r1 = r0 + Nn;
  unsigned int* o = dst + (size_t)pr * Nn + n;
  #pragma unroll
  for (int j = 0; j < 8; ++j) o[j] = pack2(r0[j], r1[j]);
}

// ---------------------------------------------------------------------------
// C[M,Nn] = A[M,Kd](bf16 row-major) * B[Kd,Nn](bf16 K-pair-packed dwords)
// OUTMODE: 0 = f32 row-major, 1 = bf16 row-major, 2 = bf16 K-pair-packed
// grid (Nn/128, M/128, batch), 256 threads = 8 wave32 (2x4)
// ---------------------------------------------------------------------------
template<int OUTMODE>
__global__ __launch_bounds__(256) void gemm_nn(
    const unsigned short* __restrict__ Ap, const unsigned int* __restrict__ Bp,
    void* __restrict__ Cp, int Nn, int Kd, long sA, long sB, long sC)
{
  __shared__ __align__(16) unsigned short ldsA[2][BM * LDA];
  __shared__ __align__(16) unsigned int   ldsB[2][(BK / 2) * BN];

  const int tid  = threadIdx.x;
  const int bz   = blockIdx.z;
  const int m0g  = blockIdx.y * BM;
  const int n0g  = blockIdx.x * BN;

  const int wave = tid >> 5, lane = tid & 31;
  const int wrow = wave >> 2, wcol = wave & 3;
  const int lm = lane & 15,  lh = lane >> 4;

  const int ar = tid >> 1,  ac   = (tid & 1) * 32;   // A: row 0..127, 32-elem seg
  const int br = tid >> 3,  bseg = (tid & 7) * 16;   // B: pair-row 0..31, 16-dword seg

  const unsigned short* agb = Ap + (size_t)bz * sA + (size_t)(m0g + ar) * Kd + ac;
  const unsigned int*   bgb = Bp + (size_t)bz * sB + (size_t)br * Nn + n0g + bseg;

  const unsigned daA0 = lds_off(&ldsA[0][ar * LDA + ac]);
  const unsigned daA1 = lds_off(&ldsA[1][ar * LDA + ac]);
  const unsigned daB0 = lds_off(&ldsB[0][br * BN + bseg]);
  const unsigned daB1 = lds_off(&ldsB[1][br * BN + bseg]);

  auto stage = [&](int kt, int buf) {
    async_copy64B(buf ? daA1 : daA0, agb + kt);
    async_copy64B(buf ? daB1 : daB0, bgb + (size_t)(kt >> 1) * Nn);
  };

  v8f acc[4][2];
  #pragma unroll
  for (int mi = 0; mi < 4; ++mi)
    #pragma unroll
    for (int ni = 0; ni < 2; ++ni)
      #pragma unroll
      for (int r = 0; r < 8; ++r) acc[mi][ni][r] = 0.0f;

  const int nk = Kd / BK;
  stage(0, 0);
  for (int t = 0; t < nk; ++t) {
    const int buf = t & 1;
    wait_async0();
    __syncthreads();
    if (t + 1 < nk) stage((t + 1) * BK, buf ^ 1);

    #pragma unroll
    for (int ks = 0; ks < BK; ks += 32) {
      ABFrag a[4], b[2];
      #pragma unroll
      for (int mi = 0; mi < 4; ++mi) {
        const int m = wrow * 64 + mi * 16 + lm;
        a[mi].q[0] = *(const U4*)&ldsA[buf][m * LDA + ks + lh * 16];
        a[mi].q[1] = *(const U4*)&ldsA[buf][m * LDA + ks + lh * 16 + 8];
      }
      #pragma unroll
      for (int ni = 0; ni < 2; ++ni) {
        const int n = wcol * 32 + ni * 16 + lm;
        const unsigned int* bp = &ldsB[buf][((ks >> 1) + lh * 8) * BN + n];
        #pragma unroll
        for (int v = 0; v < 8; ++v) b[ni].d[v] = bp[v * BN];
      }
      #pragma unroll
      for (int mi = 0; mi < 4; ++mi)
        #pragma unroll
        for (int ni = 0; ni < 2; ++ni)
          acc[mi][ni] = __builtin_amdgcn_wmma_f32_16x16x32_bf16(
              false, a[mi].v, false, b[ni].v, (short)0, acc[mi][ni], false, false);
    }
  }

  #pragma unroll
  for (int mi = 0; mi < 4; ++mi)
    #pragma unroll
    for (int ni = 0; ni < 2; ++ni) {
      const int n = n0g + wcol * 32 + ni * 16 + lm;
      const int mb = m0g + wrow * 64 + mi * 16 + lh * 8;
      if (OUTMODE == 2) {
        unsigned int* cp = (unsigned int*)Cp + (size_t)bz * sC;
        #pragma unroll
        for (int r = 0; r < 8; r += 2)
          cp[(size_t)((mb + r) >> 1) * Nn + n] = pack2(acc[mi][ni][r], acc[mi][ni][r + 1]);
      } else if (OUTMODE == 1) {
        unsigned short* cp = (unsigned short*)Cp + (size_t)bz * sC;
        #pragma unroll
        for (int r = 0; r < 8; ++r)
          cp[(size_t)(mb + r) * Nn + n] = f2bf(acc[mi][ni][r]);
      } else {
        float* cp = (float*)Cp + (size_t)bz * sC;
        #pragma unroll
        for (int r = 0; r < 8; ++r)
          cp[(size_t)(mb + r) * Nn + n] = acc[mi][ni][r];
      }
    }
}

// ---------------------------------------------------------------------------
// C[M,Nn] = A[M,Kd] * B^T, B stored [Nn, Kd] row-major (scores = Q K^T over n)
// bf16 in, f32 out.
// ---------------------------------------------------------------------------
__global__ __launch_bounds__(256) void gemm_nt(
    const unsigned short* __restrict__ Ap, const unsigned short* __restrict__ Bp,
    float* __restrict__ Cp, int Nn, int Kd, long sA, long sB, long sC)
{
  __shared__ __align__(16) unsigned short ldsA [2][BM * LDA];
  __shared__ __align__(16) unsigned short ldsBt[2][BN * LDA];

  const int tid  = threadIdx.x;
  const int bz   = blockIdx.z;
  const int m0g  = blockIdx.y * BM;
  const int n0g  = blockIdx.x * BN;

  const int wave = tid >> 5, lane = tid & 31;
  const int wrow = wave >> 2, wcol = wave & 3;
  const int lm = lane & 15,  lh = lane >> 4;

  const int ar = tid >> 1, ac = (tid & 1) * 32;

  const unsigned short* agb = Ap + (size_t)bz * sA + (size_t)(m0g + ar) * Kd + ac;
  const unsigned short* bgb = Bp + (size_t)bz * sB + (size_t)(n0g + ar) * Kd + ac;

  const unsigned daA0 = lds_off(&ldsA[0][ar * LDA + ac]);
  const unsigned daA1 = lds_off(&ldsA[1][ar * LDA + ac]);
  const unsigned daB0 = lds_off(&ldsBt[0][ar * LDA + ac]);
  const unsigned daB1 = lds_off(&ldsBt[1][ar * LDA + ac]);

  auto stage = [&](int kt, int buf) {
    async_copy64B(buf ? daA1 : daA0, agb + kt);
    async_copy64B(buf ? daB1 : daB0, bgb + kt);
  };

  v8f acc[4][2];
  #pragma unroll
  for (int mi = 0; mi < 4; ++mi)
    #pragma unroll
    for (int ni = 0; ni < 2; ++ni)
      #pragma unroll
      for (int r = 0; r < 8; ++r) acc[mi][ni][r] = 0.0f;

  const int nk = Kd / BK;
  stage(0, 0);
  for (int t = 0; t < nk; ++t) {
    const int buf = t & 1;
    wait_async0();
    __syncthreads();
    if (t + 1 < nk) stage((t + 1) * BK, buf ^ 1);

    #pragma unroll
    for (int ks = 0; ks < BK; ks += 32) {
      ABFrag a[4], b[2];
      #pragma unroll
      for (int mi = 0; mi < 4; ++mi) {
        const int m = wrow * 64 + mi * 16 + lm;
        a[mi].q[0] = *(const U4*)&ldsA[buf][m * LDA + ks + lh * 8];
        a[mi].q[1] = *(const U4*)&ldsA[buf][m * LDA + ks + 16 + lh * 8];
      }
      #pragma unroll
      for (int ni = 0; ni < 2; ++ni) {
        const int n = wcol * 32 + ni * 16 + lm;
        b[ni].q[0] = *(const U4*)&ldsBt[buf][n * LDA + ks + lh * 8];
        b[ni].q[1] = *(const U4*)&ldsBt[buf][n * LDA + ks + 16 + lh * 8];
      }
      #pragma unroll
      for (int mi = 0; mi < 4; ++mi)
        #pragma unroll
        for (int ni = 0; ni < 2; ++ni)
          acc[mi][ni] = __builtin_amdgcn_wmma_f32_16x16x32_bf16(
              false, a[mi].v, false, b[ni].v, (short)0, acc[mi][ni], false, false);
    }
  }

  #pragma unroll
  for (int mi = 0; mi < 4; ++mi)
    #pragma unroll
    for (int ni = 0; ni < 2; ++ni) {
      const int n = n0g + wcol * 32 + ni * 16 + lm;
      #pragma unroll
      for (int r = 0; r < 8; ++r) {
        const int m = m0g + wrow * 64 + mi * 16 + lh * 8 + r;
        Cp[(size_t)bz * sC + (size_t)m * Nn + n] = acc[mi][ni][r];
      }
    }
}

// ---------------------------------------------------------------------------
// softmax over q (axis=1) of S[b, q, k], scale 1/sqrt(1024)=1/32; P as bf16.
// ---------------------------------------------------------------------------
__global__ __launch_bounds__(256) void softmax_col(
    const float* __restrict__ S, unsigned short* __restrict__ P)
{
  const int b = blockIdx.x >> 2;
  const int k = ((blockIdx.x & 3) * 256) + threadIdx.x;
  const float* s = S + (size_t)b * 1024 * 1024 + k;
  unsigned short* p = P + (size_t)b * 1024 * 1024 + k;

  float mx = -3.4e38f;
  for (int q = 0; q < 1024; ++q) mx = fmaxf(mx, s[(size_t)q * 1024]);
  float sum = 0.0f;
  for (int q = 0; q < 1024; ++q) sum += __expf((s[(size_t)q * 1024] - mx) * 0.03125f);
  const float inv = 1.0f / sum;
  for (int q = 0; q < 1024; ++q)
    p[(size_t)q * 1024] = f2bf(__expf((s[(size_t)q * 1024] - mx) * 0.03125f) * inv);
}

// ---------------------------------------------------------------------------
extern "C" void kernel_launch(void* const* d_in, const int* in_sizes, int n_in,
                              void* d_out, int out_size, void* d_ws, size_t ws_size,
                              hipStream_t stream) {
  (void)in_sizes; (void)n_in; (void)out_size; (void)ws_size;
  const float* x  = (const float*)d_in[0];   // (4, 1024, 4096) f32
  const float* Wq = (const float*)d_in[1];   // (1024, 1024) f32
  const float* Wk = (const float*)d_in[2];
  const float* Wv = (const float*)d_in[3];
  float* out = (float*)d_out;                // (4, 1024, 4096) f32

  const size_t WSZ  = (size_t)1024 * 1024;          // W elems (bf16)
  const size_t XP   = (size_t)4 * 512 * 4096;       // x pair-packed dwords
  const size_t QKV  = (size_t)4 * 1024 * 4096;      // Q/K elems (bf16)
  const size_t VP   = (size_t)4 * 512 * 4096;       // V pair-packed dwords
  const size_t SSZ  = (size_t)4 * 1024 * 1024;      // scores f32 / P bf16 elems

  char* w = (char*)d_ws;
  unsigned short* Wqb = (unsigned short*)w;                 w += WSZ * 2;
  unsigned short* Wkb = (unsigned short*)w;                 w += WSZ * 2;
  unsigned short* Wvb = (unsigned short*)w;                 w += WSZ * 2;
  unsigned int*   xp  = (unsigned int*)w;                   w += XP * 4;
  unsigned short* Qb  = (unsigned short*)w;                 w += QKV * 2;
  unsigned short* Kb  = (unsigned short*)w;                 w += QKV * 2;
  unsigned int*   Vp  = (unsigned int*)w;                   w += VP * 4;
  float*          Sc  = (float*)w;                          w += SSZ * 4;
  unsigned short* Pb  = (unsigned short*)w;

  const long sX  = 1024L * 4096;  // per-batch elems: x/Q/K/out rows
  const long sXP = 512L * 4096;   // per-batch dwords: xp / Vp
  const long sS  = 1024L * 1024;  // per-batch elems: scores / P

  dim3 blk(256);

  // ---- one-shot conversions
  cvt_plain<<<dim3(512), blk, 0, stream>>>(Wq, (unsigned int*)Wqb);   // 1M f32
  cvt_plain<<<dim3(512), blk, 0, stream>>>(Wk, (unsigned int*)Wkb);
  cvt_plain<<<dim3(512), blk, 0, stream>>>(Wv, (unsigned int*)Wvb);
  cvt_pack_pairs<<<dim3(4096), blk, 0, stream>>>(x, xp, 4096);        // 2048 pair-rows

  // ---- projections (A = W bf16 batch-shared, B = xp pair-packed)
  dim3 gProj(4096 / BN, 1024 / BM, 4);
  gemm_nn<1><<<gProj, blk, 0, stream>>>(Wqb, xp, Qb, 4096, 1024, 0L, sXP, sX);
  gemm_nn<1><<<gProj, blk, 0, stream>>>(Wkb, xp, Kb, 4096, 1024, 0L, sXP, sX);
  gemm_nn<2><<<gProj, blk, 0, stream>>>(Wvb, xp, Vp, 4096, 1024, 0L, sXP, sXP);

  // ---- scores S[q,k] = sum_n Q[q,n] K[k,n]
  dim3 gS(1024 / BN, 1024 / BM, 4);
  gemm_nt<<<gS, blk, 0, stream>>>(Qb, Kb, Sc, 1024, 4096, sX, sX, sS);

  // ---- column softmax over q
  softmax_col<<<dim3(16), blk, 0, stream>>>(Sc, Pb);

  // ---- out = P @ V (B = Vp pair-packed), f32 out
  gemm_nn<0><<<gProj, blk, 0, stream>>>(Pb, Vp, out, 4096, 1024, sS, sXP, sX);
}
